// RelationNet_50861002719650
// MI455X (gfx1250) — compile-verified, matrix operand order checked by reading, not run on hardware
//
#include <hip/hip_runtime.h>

#define EPS 1e-5f

// ---- problem constants -----------------------------------------------------
constexpr int B_ = 4, Q_ = 256, S_ = 128, C_ = 256;
constexpr int NT = B_ * Q_ * S_;   // 131072 total rows
constexpr int F1 = 256;            // layer-0 out channels
constexpr int F2 = 64;             // layer-1 out channels

// ---- workspace layout (float offsets) -------------------------------------
constexpr size_t OFF_AQ   = 0;                                   // B*Q*F1
constexpr size_t OFF_AS   = OFF_AQ  + (size_t)B_ * Q_ * F1;      // B*S*F1
constexpr size_t OFF_SQ   = OFF_AS  + (size_t)B_ * S_ * F1;      // B*F1
constexpr size_t OFF_SQ2  = OFF_SQ  + (size_t)B_ * F1;
constexpr size_t OFF_SS   = OFF_SQ2 + (size_t)B_ * F1;
constexpr size_t OFF_SS2  = OFF_SS  + (size_t)B_ * F1;
constexpr size_t OFF_A0   = OFF_SS2 + (size_t)B_ * F1;           // F1
constexpr size_t OFF_C0   = OFF_A0  + F1;                        // F1
constexpr size_t OFF_SUM1 = OFF_C0  + F1;                        // F2
constexpr size_t OFF_SSQ1 = OFF_SUM1 + F2;                       // F2
constexpr size_t OFF_A1   = OFF_SSQ1 + F2;                       // F2
constexpr size_t OFF_C1   = OFF_A1   + F2;                       // F2
constexpr size_t OFF_SUM2 = OFF_C1   + F2;                       // 1
constexpr size_t OFF_SSQ2 = OFF_SUM2 + 1;                        // 1
constexpr size_t OFF_A2   = OFF_SSQ2 + 1;                        // 1
constexpr size_t OFF_C2   = OFF_A2   + 1;                        // 1
constexpr size_t OFF_Y1   = 398336;                              // NT*F2
constexpr size_t OFF_Y2   = OFF_Y1 + (size_t)NT * F2;            // NT

// ---- WMMA types ------------------------------------------------------------
typedef __attribute__((ext_vector_type(16))) __bf16 v16bf;
typedef __attribute__((ext_vector_type(8)))  float  v8f;

__device__ __forceinline__ unsigned short f2bf(float f) {
  unsigned u = __float_as_uint(f);
  u += 0x7FFFu + ((u >> 16) & 1u);          // round-to-nearest-even
  return (unsigned short)(u >> 16);
}

// ---- k0: zero the atomic accumulators --------------------------------------
__global__ void k0_zero(float* ws) {
  for (int i = threadIdx.x; i < 260; i += 256) ws[OFF_SUM1 + i] = 0.f;
}

// ---- k1: Aq = query @ W0[:, :C]^T ; As = support @ W0[:, C:]^T -------------
__global__ void k1_proj(const float* __restrict__ qry, const float* __restrict__ sup,
                        const float* __restrict__ W0, float* __restrict__ ws) {
  __shared__ float row[C_];
  const int blk = blockIdx.x, tid = threadIdx.x;     // 256 threads, tid = out channel
  const float4* W4 = (const float4*)W0;              // W0 row = 512 f32 = 128 float4
  float* Aq = ws + OFF_AQ;
  float* As = ws + OFF_AS;
  if (blk < B_ * Q_) {
    row[tid] = qry[(size_t)blk * C_ + tid];
    __syncthreads();
    float acc = 0.f;
    #pragma unroll 4
    for (int c4 = 0; c4 < 64; ++c4) {
      float4 w = W4[(size_t)tid * 128 + c4];
      acc += w.x * row[4*c4] + w.y * row[4*c4+1] + w.z * row[4*c4+2] + w.w * row[4*c4+3];
    }
    Aq[(size_t)blk * F1 + tid] = acc;
  } else {
    const int sb = blk - B_ * Q_;
    row[tid] = sup[(size_t)sb * C_ + tid];
    __syncthreads();
    float acc = 0.f;
    #pragma unroll 4
    for (int c4 = 0; c4 < 64; ++c4) {
      float4 w = W4[(size_t)tid * 128 + 64 + c4];
      acc += w.x * row[4*c4] + w.y * row[4*c4+1] + w.z * row[4*c4+2] + w.w * row[4*c4+3];
    }
    As[(size_t)sb * F1 + tid] = acc;
  }
}

// ---- k2: per-(b, channel) sums of Aq and As (deterministic, no atomics) ----
__global__ void k2_stats0(float* __restrict__ ws) {
  const int b = blockIdx.x, o = threadIdx.x;         // grid=B, 256 threads
  const float* Aq = ws + OFF_AQ;
  const float* As = ws + OFF_AS;
  float sq = 0.f, sq2 = 0.f, ss = 0.f, ss2 = 0.f;
  for (int qq = 0; qq < Q_; ++qq) {
    float v = Aq[((size_t)(b * Q_ + qq)) * F1 + o];
    sq += v; sq2 += v * v;
  }
  for (int si = 0; si < S_; ++si) {
    float v = As[((size_t)(b * S_ + si)) * F1 + o];
    ss += v; ss2 += v * v;
  }
  ws[OFF_SQ  + b * F1 + o] = sq;
  ws[OFF_SQ2 + b * F1 + o] = sq2;
  ws[OFF_SS  + b * F1 + o] = ss;
  ws[OFF_SS2 + b * F1 + o] = ss2;
}

// ---- k3: fold BN0 into per-channel scale A0 / shift C0 ---------------------
__global__ void k3_bn0(const float* __restrict__ b0, const float* __restrict__ g0,
                       const float* __restrict__ bt0, float* __restrict__ ws) {
  const int o = threadIdx.x;                         // 256 threads
  float SqA = 0.f, SsA = 0.f, E2 = 0.f;
  for (int b = 0; b < B_; ++b) {
    float sq  = ws[OFF_SQ  + b * F1 + o];
    float sq2 = ws[OFF_SQ2 + b * F1 + o];
    float ss  = ws[OFF_SS  + b * F1 + o];
    float ss2 = ws[OFF_SS2 + b * F1 + o];
    SqA += sq; SsA += ss;
    E2 += (float)S_ * sq2 + (float)Q_ * ss2 + 2.f * sq * ss;
  }
  float mQ = SqA / (float)(B_ * Q_);
  float mS = SsA / (float)(B_ * S_);
  float bb = b0[o];
  float mean = mQ + mS + bb;
  float ex2  = E2 / (float)NT + 2.f * bb * (mQ + mS) + bb * bb;
  float var  = ex2 - mean * mean;
  float A0v  = g0[o] * rsqrtf(var + EPS);
  ws[OFF_A0 + o] = A0v;
  ws[OFF_C0 + o] = (bb - mean) * A0v + bt0[o];
}

// ---- k4: fused BN0+ReLU regeneration of x0 + bf16 WMMA GEMM (K=256, N=64) --
// 128 rows/block, 4 waves, each wave owns TWO 16-row M-tiles x all 64 columns:
// per K-step the 4 shared B-fragment loads feed 8 WMMAs (vs 4 before).
__launch_bounds__(128)
__global__ void k4_gemm1(const float* __restrict__ W1, const float* __restrict__ b1,
                         float* __restrict__ ws) {
  // x tile: 128x256 bf16 = 64KB at [0,32768); W1 tile: 64x256 bf16 = 32KB at [32768,49152)
  __shared__ unsigned short lds[49152];
  const int tid  = threadIdx.x;
  const int row0 = blockIdx.x * 128;
  const float* Aq = ws + OFF_AQ;
  const float* As = ws + OFF_AS;
  const float* A0 = ws + OFF_A0;
  const float* C0 = ws + OFF_C0;
  float* y1 = ws + OFF_Y1;

  // phase 1a: regenerate x0 rows through BN0+ReLU, convert to bf16 in LDS
  for (int idx = tid; idx < 128 * 256; idx += 128) {
    int r = idx >> 8, c = idx & 255;
    int n = row0 + r;
    int b = n >> 15, qq = (n >> 7) & 255, si = n & 127;
    float v = Aq[((size_t)(b * Q_ + qq)) * F1 + c] +
              As[((size_t)(b * S_ + si)) * F1 + c];
    v = v * A0[c] + C0[c];
    v = v > 0.f ? v : 0.f;
    lds[idx] = f2bf(v);
  }
  // phase 1b: stage W1 (row-major [64][256]) as bf16
  for (int idx = tid; idx < 64 * 256; idx += 128)
    lds[32768 + idx] = f2bf(W1[idx]);
  __syncthreads();

  const int wave = tid >> 5;        // 0..3, owns rows [wave*32, wave*32+32)
  const int lane = tid & 31;
  const int half = lane >> 4;       // K-group selector within fragment
  const int r    = lane & 15;       // M for A-frag / N for B-frag / N for C

  v8f acc[2][4] = {};               // 2 M-tiles x 4 N-tiles x 8 VGPRs f32

  #pragma unroll 1
  for (int ks = 0; ks < 8; ++ks) {  // K = 256 in steps of 32
    const int k0 = ks * 32;
    union { unsigned u[8]; v16bf v; } af[2];
    #pragma unroll
    for (int tt = 0; tt < 2; ++tt) {
      #pragma unroll
      for (int vv = 0; vv < 8; ++vv) {
        // 16-bit fragment K map: kbase = (v>>2)*16 + half*8 + (v&3)*2
        int k = k0 + ((vv >> 2) << 4) + (half << 3) + ((vv & 3) << 1);
        af[tt].u[vv] = *(const unsigned*)&lds[(wave * 32 + tt * 16 + r) * 256 + k];
      }
    }
    #pragma unroll
    for (int t = 0; t < 4; ++t) {
      union { unsigned u[8]; v16bf v; } bfrag;
      #pragma unroll
      for (int vv = 0; vv < 8; ++vv) {
        int k = k0 + ((vv >> 2) << 4) + (half << 3) + ((vv & 3) << 1);
        bfrag.u[vv] = *(const unsigned*)&lds[32768 + (t * 16 + r) * 256 + k];
      }
      acc[0][t] = __builtin_amdgcn_wmma_f32_16x16x32_bf16(
          false, af[0].v, false, bfrag.v, (short)0, acc[0][t], false, false);
      acc[1][t] = __builtin_amdgcn_wmma_f32_16x16x32_bf16(
          false, af[1].v, false, bfrag.v, (short)0, acc[1][t], false, false);
    }
  }

  // phase 3: +b1, store y1, accumulate BN1 partials
  float ps[4] = {0.f, 0.f, 0.f, 0.f};
  float pq[4] = {0.f, 0.f, 0.f, 0.f};
  #pragma unroll
  for (int tt = 0; tt < 2; ++tt) {
    const int mbase = row0 + wave * 32 + tt * 16;
    #pragma unroll
    for (int t = 0; t < 4; ++t) {
      const int j = t * 16 + r;
      const float bj = b1[j];
      #pragma unroll
      for (int i = 0; i < 8; ++i) {
        int m = i + (half << 3);    // C layout: VGPR i -> M=i (lanes<16) / M=i+8
        float v = acc[tt][t][i] + bj;
        y1[(size_t)(mbase + m) * F2 + j] = v;
        ps[t] += v; pq[t] += v * v;
      }
    }
  }
  __syncthreads();                  // LDS data no longer needed; reuse for stats
  float* sh = (float*)lds;          // sh[0..63]=sum, sh[64..127]=sumsq
  if (tid < 128) sh[tid] = 0.f;
  __syncthreads();
  #pragma unroll
  for (int t = 0; t < 4; ++t) {
    const int j = t * 16 + r;
    atomicAdd(&sh[j], ps[t]);
    atomicAdd(&sh[64 + j], pq[t]);
  }
  __syncthreads();
  if (tid < 64)        atomicAdd(&ws[OFF_SUM1 + tid], sh[tid]);
  else if (tid < 128)  atomicAdd(&ws[OFF_SSQ1 + tid - 64], sh[tid]);
}

// ---- k5: finalize BN1 -------------------------------------------------------
__global__ void k5_bn1(const float* __restrict__ g1, const float* __restrict__ bt1,
                       float* __restrict__ ws) {
  const int j = threadIdx.x;                         // 64 threads
  float mean = ws[OFF_SUM1 + j] / (float)NT;
  float var  = ws[OFF_SSQ1 + j] / (float)NT - mean * mean;
  float A1   = g1[j] * rsqrtf(var + EPS);
  ws[OFF_A1 + j] = A1;
  ws[OFF_C1 + j] = bt1[j] - mean * A1;               // b1 already folded into y1
}

// ---- k6: layer 2 matvec (K=64, N=1) + BN2 partial stats ---------------------
// y1 tile is staged with CDNA5 async global->LDS copies (ASYNCcnt tracked).
__global__ void k6_layer2(const float* __restrict__ W2, const float* __restrict__ b2,
                          float* __restrict__ ws) {
  __shared__ float ty[128 * 64];                     // 32KB staged y1 tile
  __shared__ float sA1[64], sC1[64], sW2[64];
  __shared__ float part[256];
  __shared__ float bsum, bsq;
  const int tid  = threadIdx.x;
  const int row0 = blockIdx.x * 128;
  const float* y1 = ws + OFF_Y1;
  float* y2 = ws + OFF_Y2;
  if (tid < 64) { sA1[tid] = ws[OFF_A1 + tid]; sC1[tid] = ws[OFF_C1 + tid]; sW2[tid] = W2[tid]; }
  if (tid == 0) { bsum = 0.f; bsq = 0.f; }
  // async copy: 8192 floats = 2048 x b128; each lane issues an async-to-LDS op.
  const float* src = y1 + (size_t)row0 * 64;
  #pragma unroll
  for (int it = 0; it < 8; ++it) {
    int idx = tid + it * 256;                        // 16-byte chunk index
    unsigned ldsaddr = (unsigned)(uintptr_t)(&ty[idx * 4]);
    const float* g = src + idx * 4;
    asm volatile("global_load_async_to_lds_b128 %0, %1, off"
                 :: "v"(ldsaddr), "v"(g) : "memory");
  }
  asm volatile("s_wait_asynccnt 0x0" ::: "memory");
  __syncthreads();
  const int rr = tid >> 1, h = tid & 1;              // 2 threads per row
  float p = 0.f;
  #pragma unroll 8
  for (int jj = 0; jj < 32; ++jj) {
    int j = h * 32 + jj;
    float v = ty[rr * 64 + j] * sA1[j] + sC1[j];
    v = v > 0.f ? v : 0.f;
    p += v * sW2[j];
  }
  part[tid] = p;
  __syncthreads();
  if (h == 0) {
    float z = part[tid] + part[tid + 1] + b2[0];
    y2[row0 + rr] = z;
    atomicAdd(&bsum, z);
    atomicAdd(&bsq, z * z);
  }
  __syncthreads();
  if (tid == 0) { atomicAdd(&ws[OFF_SUM2], bsum); atomicAdd(&ws[OFF_SSQ2], bsq); }
}

// ---- k7: finalize BN2 (single channel) -------------------------------------
__global__ void k7_bn2(const float* __restrict__ g2, const float* __restrict__ bt2,
                       float* __restrict__ ws) {
  float mean = ws[OFF_SUM2] / (float)NT;
  float var  = ws[OFF_SSQ2] / (float)NT - mean * mean;
  float A2   = g2[0] * rsqrtf(var + EPS);
  ws[OFF_A2] = A2;
  ws[OFF_C2] = bt2[0] - mean * A2;
}

// ---- k8: apply BN2 + ReLU; [B,N,1] -> [B,1,Q,S] is the identity flatten ----
__global__ void k8_out(const float* __restrict__ ws, float* __restrict__ out) {
  const int n = blockIdx.x * 256 + threadIdx.x;
  float v = ws[OFF_Y2 + n] * ws[OFF_A2] + ws[OFF_C2];
  out[n] = v > 0.f ? v : 0.f;
}

// ---- host entry -------------------------------------------------------------
extern "C" void kernel_launch(void* const* d_in, const int* in_sizes, int n_in,
                              void* d_out, int out_size, void* d_ws, size_t ws_size,
                              hipStream_t stream) {
  const float* sup = (const float*)d_in[0];
  const float* qry = (const float*)d_in[1];
  const float* W0  = (const float*)d_in[2];
  const float* b0  = (const float*)d_in[3];
  const float* g0  = (const float*)d_in[4];
  const float* bt0 = (const float*)d_in[5];
  const float* W1  = (const float*)d_in[6];
  const float* b1  = (const float*)d_in[7];
  const float* g1  = (const float*)d_in[8];
  const float* bt1 = (const float*)d_in[9];
  const float* W2  = (const float*)d_in[10];
  const float* b2  = (const float*)d_in[11];
  const float* g2  = (const float*)d_in[12];
  const float* bt2 = (const float*)d_in[13];
  float* ws  = (float*)d_ws;
  float* out = (float*)d_out;

  k0_zero  <<<1, 256, 0, stream>>>(ws);
  k1_proj  <<<B_ * Q_ + B_ * S_, 256, 0, stream>>>(qry, sup, W0, ws);
  k2_stats0<<<B_, 256, 0, stream>>>(ws);
  k3_bn0   <<<1, 256, 0, stream>>>(b0, g0, bt0, ws);
  k4_gemm1 <<<NT / 128, 128, 0, stream>>>(W1, b1, ws);
  k5_bn1   <<<1, 64, 0, stream>>>(g1, bt1, ws);
  k6_layer2<<<NT / 128, 256, 0, stream>>>(W2, b2, ws);
  k7_bn2   <<<1, 1, 0, stream>>>(g2, bt2, ws);
  k8_out   <<<NT / 256, 256, 0, stream>>>(ws, out);
}
